// MockFP8Linear_64407329571542
// MI455X (gfx1250) — matrix operand order repaired and checked by reading, not compiled, not run on hardware
//
#include <hip/hip_runtime.h>
#include <cstdint>
#include <cstddef>

// ---------------------------------------------------------------------------
// out[t,o] = sum_i x[t,i] * W[o,i] * scale[o/128, i/128]
// M=16384, N=2048, K=2048.
// Pass 1: W * scale -> bf16 (ws).  Pass 2: X -> bf16 (ws).
// Pass 3: bf16 GEMM, v_wmma_f32_16x16x32_bf16, async global->LDS copies
//         (global_load_async_to_lds_b128, ASYNCcnt), double-buffered LDS.
// Fallback (ws too small): single-kernel inline-conversion GEMM.
// ---------------------------------------------------------------------------

typedef __attribute__((ext_vector_type(16))) __bf16 v16bf;
typedef __attribute__((ext_vector_type(8)))  __bf16 v8bf;
typedef __attribute__((ext_vector_type(2)))  __bf16 v2bf;
typedef __attribute__((ext_vector_type(8)))  float  v8f;

union V16 { v16bf v; v8bf h[2]; };

constexpr int TOKENS = 16384;
constexpr int IN_F   = 2048;
constexpr int OUT_F  = 2048;
constexpr int SB_I   = IN_F / 128;

#if defined(__has_builtin)
#if __has_builtin(__builtin_amdgcn_cvt_pk_bf16_f32)
#define HAVE_PK_BF16_CVT 1
#endif
#endif

__device__ __forceinline__ unsigned pack_bf16x2(float lo, float hi) {
#ifdef HAVE_PK_BF16_CVT
  v2bf p = __builtin_amdgcn_cvt_pk_bf16_f32(lo, hi);
  return __builtin_bit_cast(unsigned, p);
#else
  unsigned short ua = __builtin_bit_cast(unsigned short, (__bf16)lo);
  unsigned short ub = __builtin_bit_cast(unsigned short, (__bf16)hi);
  return (unsigned)ua | ((unsigned)ub << 16);
#endif
}

// ---------------- Pass 1: W * blockscale -> bf16 ----------------
__global__ __launch_bounds__(256)
void dequant_w_bf16(const float* __restrict__ W, const float* __restrict__ S,
                    unsigned short* __restrict__ Wb) {
  size_t base = ((size_t)blockIdx.x * 256 + threadIdx.x) * 8;  // 8 elems/thread
  int row = (int)(base / IN_F);
  int col = (int)(base % IN_F);                 // 8-aligned; no 128-block crossing
  float s = S[(row >> 7) * SB_I + (col >> 7)];
  float4 a = *reinterpret_cast<const float4*>(W + base);
  float4 b = *reinterpret_cast<const float4*>(W + base + 4);
  uint4 o;
  o.x = pack_bf16x2(a.x * s, a.y * s);
  o.y = pack_bf16x2(a.z * s, a.w * s);
  o.z = pack_bf16x2(b.x * s, b.y * s);
  o.w = pack_bf16x2(b.z * s, b.w * s);
  *reinterpret_cast<uint4*>(Wb + base) = o;
}

// ---------------- Pass 2: X -> bf16 ----------------
__global__ __launch_bounds__(256)
void cvt_x_bf16(const float* __restrict__ X, unsigned short* __restrict__ Xb) {
  size_t base = ((size_t)blockIdx.x * 256 + threadIdx.x) * 8;
  float4 a = *reinterpret_cast<const float4*>(X + base);
  float4 b = *reinterpret_cast<const float4*>(X + base + 4);
  uint4 o;
  o.x = pack_bf16x2(a.x, a.y);
  o.y = pack_bf16x2(a.z, a.w);
  o.z = pack_bf16x2(b.x, b.y);
  o.w = pack_bf16x2(b.z, b.w);
  *reinterpret_cast<uint4*>(Xb + base) = o;
}

// ---------------- Pass 3: bf16 GEMM with async LDS fill ----------------
#define M_TILE 128
#define N_TILE 128
#define K_TILE 64
#define LDS_STRIDE 72   // halves: 144B rows (9x16B aligned), 36-bank stagger

__device__ __forceinline__ void async_load_b128(unsigned lds_off, const void* g) {
  // gfx1250: memory -> LDS, 16B per lane, tracked by ASYNCcnt
  asm volatile("global_load_async_to_lds_b128 %0, %1, off"
               :: "v"(lds_off), "v"(g) : "memory");
}
__device__ __forceinline__ void wait_asynccnt0() {
  asm volatile("s_wait_asynccnt 0x0" ::: "memory");
}

__global__ __launch_bounds__(256)
void gemm_bf16_wmma(const unsigned short* __restrict__ Xb,
                    const unsigned short* __restrict__ Wb,
                    float* __restrict__ Out) {
  __shared__ unsigned short ldsA[2][M_TILE * LDS_STRIDE];
  __shared__ unsigned short ldsB[2][N_TILE * LDS_STRIDE];

  const int tid   = threadIdx.x;
  const int lane  = tid & 31;
  const int wave  = tid >> 5;
  const int wm    = wave & 3;     // 32-row group
  const int wn    = wave >> 2;    // 64-col group
  const int lrow  = lane & 15;
  const int khalf = lane >> 4;

  const int m0 = blockIdx.y * M_TILE;
  const int n0 = blockIdx.x * N_TILE;

  v8f acc[2][4] = {};

  // Tile = 128 rows x 64 bf16 = 128B/row = 8 x16B chunks; 1024 chunks / 256 thr = 4.
  const int c_row = tid >> 3;          // 0..31 (+32 per i)
  const int c_col = (tid & 7) * 8;     // halves within row: chunk of 8 bf16 = 16B

  // LDS byte offsets for this thread's 4 chunks (A and B use same pattern)
  unsigned ldsA0 = (unsigned)(uintptr_t)(
      (__attribute__((address_space(3))) unsigned short*)&ldsA[0][0]);
  unsigned ldsB0 = (unsigned)(uintptr_t)(
      (__attribute__((address_space(3))) unsigned short*)&ldsB[0][0]);
  const unsigned bufA = sizeof(ldsA[0]);   // bytes per A buffer
  const unsigned bufB = sizeof(ldsB[0]);

  auto issue_async = [&](int buf, int kt) {
    const int kk = kt * K_TILE;  // halves
#pragma unroll
    for (int i = 0; i < 4; ++i) {
      int r = c_row + i * 32;
      unsigned la = ldsA0 + buf * bufA + (r * LDS_STRIDE + c_col) * 2;
      async_load_b128(la, Xb + (size_t)(m0 + r) * IN_F + kk + c_col);
    }
#pragma unroll
    for (int i = 0; i < 4; ++i) {
      int r = c_row + i * 32;
      unsigned lb = ldsB0 + buf * bufB + (r * LDS_STRIDE + c_col) * 2;
      async_load_b128(lb, Wb + (size_t)(n0 + r) * IN_F + kk + c_col);
    }
  };

  auto compute = [&](int buf) {
#pragma unroll
    for (int ks = 0; ks < 2; ++ks) {   // two 32-wide K sub-steps per stage
      V16 a[2], b[4];
      // A frags (contiguous K order, same as B; not the ISA table order): lane = M row; lanes 0-15 K 0..15,
      // lanes 16-31 K 16..31.
#pragma unroll
      for (int mi = 0; mi < 2; ++mi) {
        int row = wm * 32 + mi * 16 + lrow;
        const v8bf* p = reinterpret_cast<const v8bf*>(
            &ldsA[buf][row * LDS_STRIDE + ks * 32 + khalf * 16]);
        a[mi].h[0] = p[0];
        a[mi].h[1] = p[1];   // +8 halves
      }
      // B frags: lane = N column (= W row, K-contiguous); lanes 0-15 K 0..15,
      // lanes 16-31 K 16..31.
#pragma unroll
      for (int ni = 0; ni < 4; ++ni) {
        int col = wn * 64 + ni * 16 + lrow;
        const v8bf* q = reinterpret_cast<const v8bf*>(
            &ldsB[buf][col * LDS_STRIDE + ks * 32 + khalf * 16]);
        b[ni].h[0] = q[0];
        b[ni].h[1] = q[1];
      }
#pragma unroll
      for (int mi = 0; mi < 2; ++mi)
#pragma unroll
        for (int ni = 0; ni < 4; ++ni)
          acc[mi][ni] = __builtin_amdgcn_wmma_f32_16x16x32_bf16(
              false, a[mi].v, false, b[ni].v, (short)0, acc[mi][ni], false, false);
    }
  };

  const int NK = IN_F / K_TILE;   // 32 stages

  issue_async(0, 0);
  wait_asynccnt0();
  __syncthreads();

  for (int kt = 0; kt < NK; ++kt) {
    const int cur = kt & 1;
    if (kt + 1 < NK) issue_async(cur ^ 1, kt + 1);  // fill other buffer
    compute(cur);                                   // 24 ds_load + 16 wmma
    wait_asynccnt0();                               // my async writes done
    __syncthreads();                                // everyone done
  }

  // Epilogue: 16x16 f32 C layout -> VGPR r is row r (lanes 0-15) / r+8 (16-31)
#pragma unroll
  for (int mi = 0; mi < 2; ++mi) {
#pragma unroll
    for (int ni = 0; ni < 4; ++ni) {
      int rbase = m0 + wm * 32 + mi * 16 + khalf * 8;
      int col   = n0 + wn * 64 + ni * 16 + lrow;
#pragma unroll
      for (int r = 0; r < 8; ++r)
        Out[(size_t)(rbase + r) * OUT_F + col] = acc[mi][ni][r];
    }
  }
}

// ---------------- Fallback: inline-conversion GEMM (round-2 design) --------
#define FLDS_STRIDE 40
__global__ __launch_bounds__(256)
void gemm_fallback(const float* __restrict__ X, const float* __restrict__ W,
                   const float* __restrict__ S, float* __restrict__ Out) {
  __shared__ unsigned short ldsA[2][128 * FLDS_STRIDE];
  __shared__ unsigned short ldsB[2][128 * FLDS_STRIDE];

  const int tid = threadIdx.x, lane = tid & 31, wave = tid >> 5;
  const int wm = wave & 3, wn = wave >> 2, lrow = lane & 15, khalf = lane >> 4;
  const int m0 = blockIdx.y * 128, n0 = blockIdx.x * 128;
  v8f acc[2][4] = {};
  float4 ra[4], rb[4];
  const int g_r = tid >> 3, g_c = (tid & 7) << 2;

  auto g_load = [&](int kt) {
    const int kk = kt * 32;
#pragma unroll
    for (int i = 0; i < 4; ++i)
      ra[i] = *reinterpret_cast<const float4*>(X + (size_t)(m0 + g_r + i * 32) * IN_F + kk + g_c);
#pragma unroll
    for (int i = 0; i < 4; ++i)
      rb[i] = *reinterpret_cast<const float4*>(W + (size_t)(n0 + g_r + i * 32) * IN_F + kk + g_c);
  };
  auto l_store = [&](int buf, float s) {
#pragma unroll
    for (int i = 0; i < 4; ++i) {
      unsigned* p = reinterpret_cast<unsigned*>(&ldsA[buf][(g_r + i * 32) * FLDS_STRIDE + g_c]);
      p[0] = pack_bf16x2(ra[i].x, ra[i].y);
      p[1] = pack_bf16x2(ra[i].z, ra[i].w);
    }
#pragma unroll
    for (int i = 0; i < 4; ++i) {
      unsigned* p = reinterpret_cast<unsigned*>(&ldsB[buf][(g_r + i * 32) * FLDS_STRIDE + g_c]);
      p[0] = pack_bf16x2(rb[i].x * s, rb[i].y * s);
      p[1] = pack_bf16x2(rb[i].z * s, rb[i].w * s);
    }
  };
  auto compute = [&](int buf) {
    V16 a[2], b[4];
#pragma unroll
    for (int mi = 0; mi < 2; ++mi) {
      int row = wm * 32 + mi * 16 + lrow;
      const v8bf* p = reinterpret_cast<const v8bf*>(&ldsA[buf][row * FLDS_STRIDE + khalf * 16]);
      a[mi].h[0] = p[0]; a[mi].h[1] = p[1];
    }
#pragma unroll
    for (int ni = 0; ni < 4; ++ni) {
      int col = wn * 64 + ni * 16 + lrow;
      const v8bf* q = reinterpret_cast<const v8bf*>(&ldsB[buf][col * FLDS_STRIDE + khalf * 16]);
      b[ni].h[0] = q[0]; b[ni].h[1] = q[1];
    }
#pragma unroll
    for (int mi = 0; mi < 2; ++mi)
#pragma unroll
      for (int ni = 0; ni < 4; ++ni)
        acc[mi][ni] = __builtin_amdgcn_wmma_f32_16x16x32_bf16(
            false, a[mi].v, false, b[ni].v, (short)0, acc[mi][ni], false, false);
  };

  const int NK = IN_F / 32;
  const int so = (n0 >> 7) * SB_I;
  g_load(0); l_store(0, S[so]); __syncthreads();
  for (int kt = 0; kt < NK; ++kt) {
    const int cur = kt & 1;
    const bool have_next = (kt + 1) < NK;
    if (have_next) g_load(kt + 1);
    compute(cur);
    if (have_next) l_store(cur ^ 1, S[so + ((kt + 1) >> 2)]);
    __syncthreads();
  }
#pragma unroll
  for (int mi = 0; mi < 2; ++mi)
#pragma unroll
    for (int ni = 0; ni < 4; ++ni) {
      int rbase = m0 + wm * 32 + mi * 16 + khalf * 8;
      int col   = n0 + wn * 64 + ni * 16 + lrow;
#pragma unroll
      for (int r = 0; r < 8; ++r)
        Out[(size_t)(rbase + r) * OUT_F + col] = acc[mi][ni][r];
    }
}

extern "C" void kernel_launch(void* const* d_in, const int* in_sizes, int n_in,
                              void* d_out, int out_size, void* d_ws, size_t ws_size,
                              hipStream_t stream) {
  const float* x = (const float*)d_in[0];
  const float* w = (const float*)d_in[1];
  const float* s = (const float*)d_in[2];
  float* out = (float*)d_out;

  const size_t wb_bytes = (size_t)OUT_F * IN_F * 2;           // 8 MB
  const size_t xb_bytes = (size_t)TOKENS * IN_F * 2;          // 64 MB

  if (ws_size >= wb_bytes + xb_bytes) {
    unsigned short* wb = (unsigned short*)d_ws;
    unsigned short* xb = (unsigned short*)((char*)d_ws + wb_bytes);

    dequant_w_bf16<<<(OUT_F * (size_t)IN_F) / (256 * 8), 256, 0, stream>>>(w, s, wb);
    cvt_x_bf16<<<(TOKENS * (size_t)IN_F) / (256 * 8), 256, 0, stream>>>(x, xb);

    dim3 grid(OUT_F / N_TILE, TOKENS / M_TILE);   // 16 x 128
    gemm_bf16_wmma<<<grid, 256, 0, stream>>>(xb, wb, out);
  } else {
    dim3 grid(OUT_F / 128, TOKENS / 128);
    gemm_fallback<<<grid, 256, 0, stream>>>(x, w, s, out);
  }
}
